// ImageCaptioner_88630945120816
// MI455X (gfx1250) — matrix operand order repaired and checked by reading, not compiled, not run on hardware
//
#include <hip/hip_runtime.h>
#include <hip/hip_bf16.h>
#include <cstdint>
#include <cstddef>

typedef __attribute__((ext_vector_type(16))) _Float16 v16h;
typedef __attribute__((ext_vector_type(8)))  _Float16 v8h;
typedef __attribute__((ext_vector_type(8)))  float    v8f;

__device__ inline v8f wmma_f16(v16h a, v16h b, v8f c) {
  // 8 args: (neg_a, A, neg_b, B, c_mod, C, reuse_a, reuse_b)
  return __builtin_amdgcn_wmma_f32_16x16x32_f16(false, a, false, b, (short)0, c, false, false);
}

// ---------------- utility kernels ----------------

__global__ void k_cast_f16(const float* __restrict__ s, _Float16* __restrict__ d, size_t n) {
  size_t i = (size_t)blockIdx.x * blockDim.x + threadIdx.x;
  size_t st = (size_t)gridDim.x * blockDim.x;
  for (; i < n; i += st) d[i] = (_Float16)s[i];
}

__global__ void k_zero_f32(float* p, int n) {
  int i = blockIdx.x * blockDim.x + threadIdx.x;
  if (i < n) p[i] = 0.f;
}
__global__ void k_zero_f16(_Float16* p, int n) {
  int i = blockIdx.x * blockDim.x + threadIdx.x;
  if (i < n) p[i] = (_Float16)0.f;
}

__global__ void k_pack_wcat(const float* __restrict__ wih, const float* __restrict__ whh,
                            _Float16* __restrict__ wcat) {
  int i = blockIdx.x * blockDim.x + threadIdx.x;
  if (i < 2048 * 1024) {
    int n = i >> 10, k = i & 1023;
    float v = (k < 512) ? wih[n * 512 + k] : whh[n * 512 + (k - 512)];
    wcat[i] = (_Float16)v;
  }
}

__global__ void k_bsum(const float* __restrict__ a, const float* __restrict__ b,
                       float* __restrict__ o) {
  int i = blockIdx.x * blockDim.x + threadIdx.x;
  if (i < 2048) o[i] = a[i] + b[i];
}

// ---------------- implicit-GEMM conv (WMMA f16 -> f32) ----------------
// GEMM: M=Cout, K=Cin*9, N=8*H*W.
// 8-wave WG computes a 64x64 tile; each wave: 1 A-frag, 2 B-frags, 2 accumulators.
// LDS layouts are fragment-contiguous so every DS access is a b128.
__global__ __launch_bounds__(256) void k_conv_wmma(
    const _Float16* __restrict__ X, const _Float16* __restrict__ Wf,
    const float* __restrict__ bias, float* __restrict__ Y,
    int Cin, int Cout, int H, int Wd) {
  const int HW = H * Wd;
  const int Npix = 8 * HW;
  const int K = Cin * 9;

  __shared__ _Float16 As[64][40];   // [m][k], 80B pitch -> 16B-aligned segments
  __shared__ _Float16 Bs[64][40];   // [n][k]  (transposed: k-contiguous per column)

  const int tid = threadIdx.x;
  const int lane = tid & 31;
  const int wv = tid >> 5;
  const int m0 = blockIdx.y * 64;
  const int n0 = blockIdx.x * 64;
  const int subM = (wv >> 1) * 16;  // 0,16,32,48
  const int subN = (wv & 1) * 32;   // 0,32

  // A staging: thread -> (row = tid>>2, 8-elem k-segment at (tid&3)*8)
  const int ar = tid >> 2;
  const int ac = (tid & 3) * 8;

  // B staging: thread owns one output column and an 8-elem k-segment
  const int bc = tid & 63;
  const int bs0 = (tid >> 6) * 8;
  const int gn = n0 + bc;
  const bool nok = gn < Npix;
  int nb = 0, nh = 0, nw = 0;
  if (nok) { nb = gn / HW; int hw = gn % HW; nh = hw / Wd; nw = hw % Wd; }

  v8f acc0 = {}, acc1 = {};
  for (int k0 = 0; k0 < K; k0 += 32) {
    // ---- stage A: one v8h global load + one b128 LDS store per thread ----
    {
      v8h av;
      const size_t base = (size_t)(m0 + ar) * K + k0 + ac;
      if (k0 + ac + 7 < K) {
        av = *(const v8h*)(Wf + base);
        __builtin_prefetch(Wf + base + 32, 0, 0);
      } else {
        for (int i = 0; i < 8; ++i)
          av[i] = (k0 + ac + i < K) ? Wf[base + i] : (_Float16)0.f;
      }
      *(v8h*)&As[ar][ac] = av;
    }
    // ---- stage B: im2col gather of 8 consecutive k, one b128 LDS store ----
    {
      v8h bv;
      for (int i = 0; i < 8; ++i) {
        int gk = k0 + bs0 + i;
        _Float16 v = (_Float16)0.f;
        if (nok && gk < K) {
          int ci = gk / 9, rs = gk % 9;
          int y = nh + rs / 3 - 1, xx = nw + rs % 3 - 1;
          if ((unsigned)y < (unsigned)H && (unsigned)xx < (unsigned)Wd)
            v = X[((size_t)(nb * Cin + ci) * H + y) * Wd + xx];
        }
        bv[i] = v;
      }
      *(v8h*)&Bs[bc][bs0] = bv;
    }
    __syncthreads();

    // ---- fragments (ISA 7.12.2 16-bit layouts), all b128 DS loads ----
    v16h a, b0, b1;
    const int rowA = subM + (lane & 15);
    const int klo = (lane < 16) ? 0 : 8;
    {
      v8h x0 = *(const v8h*)&As[rowA][klo];
      v8h x1 = *(const v8h*)&As[rowA][klo + 16];
      for (int i = 0; i < 8; ++i) { a[i] = x0[i]; a[8 + i] = x1[i]; }
    }
    const int colB = subN + (lane & 15);
    const int kl2 = (lane < 16) ? 0 : 8;
    {
      v8h x0 = *(const v8h*)&Bs[colB][kl2];
      v8h x1 = *(const v8h*)&Bs[colB][kl2 + 16];
      for (int i = 0; i < 8; ++i) { b0[i] = x0[i]; b0[8 + i] = x1[i]; }
      v8h y0 = *(const v8h*)&Bs[colB + 16][kl2];
      v8h y1 = *(const v8h*)&Bs[colB + 16][kl2 + 16];
      for (int i = 0; i < 8; ++i) { b1[i] = y0[i]; b1[8 + i] = y1[i]; }
    }
    acc0 = wmma_f16(a, b0, acc0);   // independent accumulators pipeline
    acc1 = wmma_f16(a, b1, acc1);   // through the WMMA hazard window
    __syncthreads();
  }

  for (int half = 0; half < 2; ++half) {
    v8f acc = half ? acc1 : acc0;
    int n = n0 + subN + half * 16 + (lane & 15);
    if (n < Npix) {
      int b = n / HW, hw = n % HW;
      for (int v = 0; v < 8; ++v) {
        int m = m0 + subM + v + ((lane >> 4) << 3);
        Y[(size_t)(b * Cout + m) * HW + hw] = acc[v] + bias[m];
      }
    }
  }
}

// ---------------- BN stats / apply ----------------
__global__ __launch_bounds__(256) void k_bn_stats(const float* __restrict__ Y,
                                                  float* __restrict__ mv, int C, int HW) {
  const int c = blockIdx.x;
  const int Npix = 8 * HW;
  float s = 0.f, s2 = 0.f;
  for (int i = threadIdx.x; i < Npix; i += 256) {
    int b = i / HW, hw = i % HW;
    float v = Y[(size_t)(b * C + c) * HW + hw];
    s += v; s2 += v * v;
  }
  __shared__ float sh[256], sh2[256];
  sh[threadIdx.x] = s; sh2[threadIdx.x] = s2;
  __syncthreads();
  for (int o = 128; o > 0; o >>= 1) {
    if (threadIdx.x < o) { sh[threadIdx.x] += sh[threadIdx.x + o]; sh2[threadIdx.x] += sh2[threadIdx.x + o]; }
    __syncthreads();
  }
  if (threadIdx.x == 0) {
    float mean = sh[0] / (float)Npix;
    mv[c] = mean;
    mv[C + c] = sh2[0] / (float)Npix - mean * mean;
  }
}

__global__ void k_bn_relu_f16(const float* __restrict__ Y, const float* __restrict__ mv,
                              const float* __restrict__ gamma, const float* __restrict__ beta,
                              _Float16* __restrict__ o, int C, int HW) {
  size_t total = (size_t)8 * C * HW;
  size_t i = (size_t)blockIdx.x * blockDim.x + threadIdx.x;
  size_t st = (size_t)gridDim.x * blockDim.x;
  for (; i < total; i += st) {
    int c = (int)((i / HW) % C);
    float v = (Y[i] - mv[c]) * rsqrtf(mv[C + c] + 1e-5f) * gamma[c] + beta[c];
    o[i] = (_Float16)(v > 0.f ? v : 0.f);
  }
}

__global__ void k_maxpool_f16(const _Float16* __restrict__ in, _Float16* __restrict__ o,
                              int C, int Hin, int Win) {
  int Ho = Hin >> 1, Wo = Win >> 1;
  size_t total = (size_t)8 * C * Ho * Wo;
  size_t i = (size_t)blockIdx.x * blockDim.x + threadIdx.x;
  size_t st = (size_t)gridDim.x * blockDim.x;
  for (; i < total; i += st) {
    int w = (int)(i % Wo);
    int h = (int)((i / Wo) % Ho);
    int c = (int)((i / ((size_t)Wo * Ho)) % C);
    int b = (int)(i / ((size_t)Wo * Ho * C));
    size_t base = ((size_t)(b * C + c) * Hin + 2 * h) * Win + 2 * w;
    float m = (float)in[base];
    float t = (float)in[base + 1];        if (t > m) m = t;
    t = (float)in[base + Win];            if (t > m) m = t;
    t = (float)in[base + Win + 1];        if (t > m) m = t;
    o[i] = (_Float16)m;
  }
}

// ---------------- generic 8xN = 8xK @ (NxK)^T WMMA GEMM ----------------
// One wave per 32-col tile (2 accumulators reuse one A fragment; M padded 8->16).
__global__ __launch_bounds__(128) void k_fc_wmma(
    const _Float16* __restrict__ Xf, const _Float16* __restrict__ Wf,
    const float* __restrict__ bias, const float* __restrict__ mask, float dropscale,
    int do_relu, int N, int K, float* __restrict__ out, _Float16* __restrict__ out_h) {
  const int lane = threadIdx.x & 31;
  const int wv = threadIdx.x >> 5;
  const int n0 = (blockIdx.x * 4 + wv) * 32;
  const int mrow = lane & 15;
  const bool aOk = mrow < 8;
  const int klo = (lane < 16) ? 0 : 16;
  const int kl2 = (lane < 16) ? 0 : 16;
  const int nc0 = n0 + (lane & 15);
  const int nc1 = nc0 + 16;
  const bool ok0 = nc0 < N;
  const bool ok1 = nc1 < N;

  v8f acc0 = {}, acc1 = {};
  for (int k0 = 0; k0 < K; k0 += 32) {
    v16h a = {};
    if (aOk) {
      v8h a0 = *(const v8h*)(Xf + (size_t)mrow * K + k0 + klo);
      v8h a1 = *(const v8h*)(Xf + (size_t)mrow * K + k0 + klo + 8);
      for (int i = 0; i < 8; ++i) { a[i] = a0[i]; a[8 + i] = a1[i]; }
    }
    v16h b0 = {}, b1 = {};
    if (ok0) {
      b0 = *(const v16h*)(Wf + (size_t)nc0 * K + k0 + kl2);
      __builtin_prefetch(Wf + (size_t)nc0 * K + k0 + kl2 + 256, 0, 0);
    }
    if (ok1) {
      b1 = *(const v16h*)(Wf + (size_t)nc1 * K + k0 + kl2);
      __builtin_prefetch(Wf + (size_t)nc1 * K + k0 + kl2 + 256, 0, 0);
    }
    acc0 = wmma_f16(a, b0, acc0);   // EXEC all-1 here (guards reconverged)
    acc1 = wmma_f16(a, b1, acc1);
  }

  for (int half = 0; half < 2; ++half) {
    int nc = half ? nc1 : nc0;
    if (nc >= N) continue;
    v8f acc = half ? acc1 : acc0;
    float bb = bias[nc];
    for (int v = 0; v < 8; ++v) {
      int m = v + ((lane >> 4) << 3);
      if (m < 8) {
        float val = acc[v] + bb;
        if (do_relu) val = val > 0.f ? val : 0.f;
        if (mask) val *= mask[(size_t)m * N + nc] * dropscale;
        if (out) out[(size_t)m * N + nc] = val;
        if (out_h) out_h[(size_t)m * N + nc] = (_Float16)val;
      }
    }
  }
}

// ---------------- decoder helpers ----------------
__global__ void k_feats(const float* __restrict__ fc3o, float* __restrict__ feats) {
  int i = blockIdx.x * blockDim.x + threadIdx.x;
  if (i < 8 * 512) {
    int b = i >> 9, e = i & 511;
    float s = 0.f;
    for (int j = 0; j < 8; ++j) s += fc3o[b * 4096 + j * 512 + e];
    feats[i] = s * 0.125f;
  }
}

__global__ void k_xcat(const float* __restrict__ emb, const int* __restrict__ cap,
                       const float* __restrict__ feats, const _Float16* __restrict__ hh,
                       _Float16* __restrict__ xcat, int step) {
  int i = blockIdx.x * blockDim.x + threadIdx.x;
  if (i < 8 * 1024) {
    int b = i >> 10, k = i & 1023;
    float v;
    if (k < 512)
      v = (step == 0) ? feats[b * 512 + k] : emb[(size_t)cap[b * 20 + step] * 512 + k];
    else
      v = (float)hh[b * 512 + (k - 512)];
    xcat[i] = (_Float16)v;
  }
}

__global__ void k_lstm_cell(const float* __restrict__ gates, float* __restrict__ h,
                            float* __restrict__ c, const float* __restrict__ mask,
                            _Float16* __restrict__ hh, _Float16* __restrict__ hd) {
  int i = blockIdx.x * blockDim.x + threadIdx.x;
  if (i < 8 * 512) {
    int b = i >> 9, j = i & 511;
    const float* g = gates + b * 2048;
    float ig = 1.f / (1.f + expf(-g[j]));
    float fg = 1.f / (1.f + expf(-g[512 + j]));
    float gg = tanhf(g[1024 + j]);
    float og = 1.f / (1.f + expf(-g[1536 + j]));
    float cn = fg * c[i] + ig * gg;
    float hn = og * tanhf(cn);
    c[i] = cn; h[i] = hn;
    hh[i] = (_Float16)hn;
    hd[i] = (_Float16)(hn * mask[i] * 2.f);   // / (1 - 0.5)
  }
}

// ---------------- host orchestration ----------------
extern "C" void kernel_launch(void* const* d_in, const int* in_sizes, int n_in,
                              void* d_out, int out_size, void* d_ws, size_t ws_size,
                              hipStream_t stream) {
  (void)in_sizes; (void)n_in; (void)out_size; (void)ws_size;
  const float* x        = (const float*)d_in[0];
  const int*   captions = (const int*)d_in[1];
  const float* fc1_w = (const float*)d_in[54];
  const float* fc1_b = (const float*)d_in[55];
  const float* fc2_w = (const float*)d_in[56];
  const float* fc2_b = (const float*)d_in[57];
  const float* fc3_w = (const float*)d_in[58];
  const float* fc3_b = (const float*)d_in[59];
  const float* embed = (const float*)d_in[60];
  const float* w_ih  = (const float*)d_in[61];
  const float* w_hh  = (const float*)d_in[62];
  const float* b_ih  = (const float*)d_in[63];
  const float* b_hh  = (const float*)d_in[64];
  const float* dec_w = (const float*)d_in[65];
  const float* dec_b = (const float*)d_in[66];
  const float* drop1 = (const float*)d_in[67];
  const float* drop2 = (const float*)d_in[68];
  const float* lmask = (const float*)d_in[69];
  float* out = (float*)d_out;

  char* ws = (char*)d_ws;
  size_t off = 0;
  auto alloc = [&](size_t bytes) -> char* {
    char* p = ws + off;
    off += (bytes + 255) & ~(size_t)255;
    return p;
  };

  const size_t ACT = (size_t)8 * 64 * 224 * 224;
  _Float16* actA = (_Float16*)alloc(ACT * 2);
  _Float16* actB = (_Float16*)alloc(ACT * 2);
  float*    ctmp = (float*)alloc(ACT * 4);

  struct CL { int cin, cout, hw, pool; };
  static const CL L[13] = {
    {3, 64, 224, 0}, {64, 64, 224, 1},
    {64, 128, 112, 0}, {128, 128, 112, 1},
    {128, 256, 56, 0}, {256, 256, 56, 0}, {256, 256, 56, 1},
    {256, 512, 28, 0}, {512, 512, 28, 0}, {512, 512, 28, 1},
    {512, 512, 14, 0}, {512, 512, 14, 0}, {512, 512, 14, 1}};

  size_t wtot = 0;
  for (int l = 0; l < 13; ++l) wtot += (size_t)L[l].cout * L[l].cin * 9;
  _Float16* wconv  = (_Float16*)alloc(wtot * 2);
  _Float16* fc1hw  = (_Float16*)alloc((size_t)4096 * 25088 * 2);
  _Float16* fc2hw  = (_Float16*)alloc((size_t)4096 * 4096 * 2);
  _Float16* fc3hw  = (_Float16*)alloc((size_t)4096 * 4096 * 2);
  _Float16* dechw  = (_Float16*)alloc((size_t)10000 * 512 * 2);
  _Float16* wcat   = (_Float16*)alloc((size_t)2048 * 1024 * 2);
  float* mv    = (float*)alloc(1024 * 4);
  float* fco   = (float*)alloc(8 * 4096 * 4);
  float* fc3o  = (float*)alloc(8 * 4096 * 4);
  _Float16* fc1a = (_Float16*)alloc(8 * 4096 * 2);
  _Float16* fc2a = (_Float16*)alloc(8 * 4096 * 2);
  float* feats = (float*)alloc(8 * 512 * 4);
  float* gates = (float*)alloc(8 * 2048 * 4);
  float* hbuf  = (float*)alloc(8 * 512 * 4);
  float* cbuf  = (float*)alloc(8 * 512 * 4);
  _Float16* hh = (_Float16*)alloc(8 * 512 * 2);
  _Float16* hd = (_Float16*)alloc(8 * 512 * 2);
  _Float16* xcat = (_Float16*)alloc(8 * 1024 * 2);
  float* bsum = (float*)alloc(2048 * 4);

  auto cast = [&](const float* s, _Float16* d, size_t n) {
    unsigned g = (unsigned)((n + 255) / 256);
    k_cast_f16<<<dim3(g), dim3(256), 0, stream>>>(s, d, n);
  };

  // ---- weight / input conversions (f32 -> f16 in workspace) ----
  cast(x, actA, (size_t)8 * 3 * 224 * 224);
  const _Float16* wptr[13];
  {
    size_t o2 = 0;
    for (int l = 0; l < 13; ++l) {
      size_t n = (size_t)L[l].cout * L[l].cin * 9;
      cast((const float*)d_in[2 + 4 * l], wconv + o2, n);
      wptr[l] = wconv + o2;
      o2 += n;
    }
  }
  cast(fc1_w, fc1hw, (size_t)4096 * 25088);
  cast(fc2_w, fc2hw, (size_t)4096 * 4096);
  cast(fc3_w, fc3hw, (size_t)4096 * 4096);
  cast(dec_w, dechw, (size_t)10000 * 512);
  k_pack_wcat<<<dim3((2048 * 1024 + 255) / 256), dim3(256), 0, stream>>>(w_ih, w_hh, wcat);
  k_bsum<<<dim3(8), dim3(256), 0, stream>>>(b_ih, b_hh, bsum);

  // ---- VGG conv chain ----
  _Float16* cur = actA; _Float16* nxt = actB;
  for (int l = 0; l < 13; ++l) {
    int hw = L[l].hw, HW = hw * hw, Npix = 8 * HW;
    dim3 g((Npix + 63) / 64, L[l].cout / 64);
    k_conv_wmma<<<g, dim3(256), 0, stream>>>(cur, wptr[l], (const float*)d_in[3 + 4 * l],
                                             ctmp, L[l].cin, L[l].cout, hw, hw);
    k_bn_stats<<<dim3(L[l].cout), dim3(256), 0, stream>>>(ctmp, mv, L[l].cout, HW);
    size_t tot = (size_t)8 * L[l].cout * HW;
    k_bn_relu_f16<<<dim3((unsigned)((tot + 255) / 256)), dim3(256), 0, stream>>>(
        ctmp, mv, (const float*)d_in[4 + 4 * l], (const float*)d_in[5 + 4 * l],
        nxt, L[l].cout, HW);
    { _Float16* t = cur; cur = nxt; nxt = t; }
    if (L[l].pool) {
      size_t pt = (size_t)8 * L[l].cout * (hw / 2) * (hw / 2);
      k_maxpool_f16<<<dim3((unsigned)((pt + 255) / 256)), dim3(256), 0, stream>>>(
          cur, nxt, L[l].cout, hw, hw);
      { _Float16* t = cur; cur = nxt; nxt = t; }
    }
  }
  // cur: 8 x 25088 f16 (NCHW flatten == reshape(B,-1))

  // ---- FC head ----
  k_fc_wmma<<<dim3(32), dim3(128), 0, stream>>>(cur,  fc1hw, fc1_b, drop1, 2.f, 1, 4096, 25088, fco,  fc1a);
  k_fc_wmma<<<dim3(32), dim3(128), 0, stream>>>(fc1a, fc2hw, fc2_b, drop2, 2.f, 1, 4096, 4096,  fco,  fc2a);
  k_fc_wmma<<<dim3(32), dim3(128), 0, stream>>>(fc2a, fc3hw, fc3_b, nullptr, 1.f, 0, 4096, 4096, fc3o, nullptr);
  k_feats<<<dim3(16), dim3(256), 0, stream>>>(fc3o, feats);

  // ---- LSTM decoder ----
  k_zero_f32<<<dim3(16), dim3(256), 0, stream>>>(hbuf, 8 * 512);
  k_zero_f32<<<dim3(16), dim3(256), 0, stream>>>(cbuf, 8 * 512);
  k_zero_f16<<<dim3(16), dim3(256), 0, stream>>>(hh, 8 * 512);

  for (int t = 0; t < 20; ++t) {
    k_xcat<<<dim3(32), dim3(256), 0, stream>>>(embed, captions, feats, hh, xcat, t);
    k_fc_wmma<<<dim3(16), dim3(128), 0, stream>>>(xcat, wcat, bsum, nullptr, 1.f, 0,
                                                  2048, 1024, gates, nullptr);
    k_lstm_cell<<<dim3(16), dim3(256), 0, stream>>>(gates, hbuf, cbuf, lmask, hh, hd);
    k_fc_wmma<<<dim3(79), dim3(128), 0, stream>>>(hd, dechw, dec_b, nullptr, 1.f, 0,
                                                  10000, 512, out + (size_t)t * 80000,
                                                  nullptr);
  }
}
